// CausalMultiHeadAttention_1735166788541
// MI455X (gfx1250) — matrix operand order repaired and checked by reading, not compiled, not run on hardware
//
#include <hip/hip_runtime.h>
#include <hip/hip_bf16.h>
#include <stdint.h>

typedef _Float16 v16h __attribute__((ext_vector_type(16)));
typedef _Float16 v8h  __attribute__((ext_vector_type(8)));
typedef _Float16 v4h  __attribute__((ext_vector_type(4)));
typedef float    v8f  __attribute__((ext_vector_type(8)));
typedef int      v4i  __attribute__((ext_vector_type(4)));

#define B_  32
#define L_  1024
#define H_  128
#define NH_ 2
#define D_  64
#define NEGV (-4294967295.0f)

// Async global->LDS (ASYNCcnt) path, probe-guarded: falls back to sync loads.
#if defined(__has_builtin)
#  if __has_builtin(__builtin_amdgcn_global_load_async_to_lds_b128) && \
      __has_builtin(__builtin_amdgcn_s_wait_asynccnt)
#    define USE_ASYNC_LDS 1
#  endif
#endif
#ifndef USE_ASYNC_LDS
#  define USE_ASYNC_LDS 0
#endif

// Builtin signature (from hipcc diagnostic): (v4i as1*, v4i as3*, imm int, imm int)
typedef __attribute__((address_space(1))) v4i* gas_ptr;
typedef __attribute__((address_space(3))) v4i* las_ptr;

static __device__ __forceinline__ gas_ptr to_global_v4i(const _Float16* p) {
  return (gas_ptr)const_cast<_Float16*>(p);
}
static __device__ __forceinline__ las_ptr to_lds_v4i(_Float16* p) {
  return (las_ptr)p;
}

union Frag16 { v16h v; v8h h[2]; };

static __device__ __forceinline__ v8f wmma_f32_f16(const Frag16& a, const Frag16& b, v8f c) {
  // D = A(16x32 f16) * B(32x16 f16) + C(16x16 f32)
  return __builtin_amdgcn_wmma_f32_16x16x32_f16(false, a.v, false, b.v, (short)0, c, false, false);
}

// ---------------------------------------------------------------------------
// Kernel A: QKV projections.  grid=(512, 3), block=128 (4 waves).
// y==0: Q = queries@Qw^T+Qb -> Qh[bh][l][d] f16
// y==1: K = keys@Kw^T+Kb    -> Kh[bh][l][d] f16
// y==2: V = keys@Vw^T+Vb    -> Vt[bh][d][l] f16 (transposed, direct from C-frag)
// ---------------------------------------------------------------------------
__global__ __launch_bounds__(128)
void qkv_proj_kernel(const float* __restrict__ qin, const float* __restrict__ kin,
                     const float* __restrict__ Qw, const float* __restrict__ Qb,
                     const float* __restrict__ Kw, const float* __restrict__ Kb,
                     const float* __restrict__ Vw, const float* __restrict__ Vb,
                     _Float16* __restrict__ Qh, _Float16* __restrict__ Kh,
                     _Float16* __restrict__ Vt)
{
  __shared__ __align__(16) _Float16 sW[H_ * H_];        // [n][k] == B^T, 32KB
  __shared__ __align__(16) _Float16 sX[64 * H_];        // [m][k], 16KB
  __shared__ __align__(16) _Float16 sStage[4][16 * H_]; // per-wave transpose stage

  const int tid  = threadIdx.x;
  const int proj = blockIdx.y;
  const float* x    = (proj == 0) ? qin : kin;
  const float* W    = (proj == 0) ? Qw : (proj == 1) ? Kw : Vw;
  const float* bias = (proj == 0) ? Qb : (proj == 1) ? Kb : Vb;

  for (int i = tid; i < H_ * H_ / 4; i += 128) {
    float4 w = ((const float4*)W)[i];
    v4h h; h[0] = (_Float16)w.x; h[1] = (_Float16)w.y; h[2] = (_Float16)w.z; h[3] = (_Float16)w.w;
    *(v4h*)&sW[i * 4] = h;
  }
  const int gm0 = blockIdx.x * 64;
  for (int i = tid; i < 64 * H_ / 4; i += 128) {
    float4 w = ((const float4*)(x + (size_t)gm0 * H_))[i];
    v4h h; h[0] = (_Float16)w.x; h[1] = (_Float16)w.y; h[2] = (_Float16)w.z; h[3] = (_Float16)w.w;
    *(v4h*)&sX[i * 4] = h;
  }
  __syncthreads();

  const int wv = tid >> 5, lane = tid & 31;
  const int lh = lane >> 4, ln = lane & 15;
  const int m0 = wv * 16;

  // A fragments (16 rows x 128 k), reused across all 8 n-tiles
  Frag16 a[4];
  #pragma unroll
  for (int kt = 0; kt < 4; ++kt) {
    int base = (m0 + ln) * H_ + kt * 32 + lh * 8;
    a[kt].h[0] = *(const v8h*)&sX[base];
    a[kt].h[1] = *(const v8h*)&sX[base + 16];
  }

  const int b_idx = gm0 >> 10;
  const int l0 = gm0 & (L_ - 1);

  #pragma unroll
  for (int nt = 0; nt < 8; ++nt) {
    v8f c = {};
    #pragma unroll
    for (int kt = 0; kt < 4; ++kt) {
      Frag16 bfr;
      int bb = (nt * 16 + ln) * H_ + kt * 32 + lh * 8;
      bfr.h[0] = *(const v8h*)&sW[bb];
      bfr.h[1] = *(const v8h*)&sW[bb + 16];
      c = wmma_f32_f16(a[kt], bfr, c);
    }
    float bv = bias[nt * 16 + ln];
    #pragma unroll
    for (int j = 0; j < 8; ++j) c[j] += bv;

    if (proj == 2) {
      // transposed store: lane's 8 accum rows = 8 consecutive l -> one b128
      int head = nt >> 2;
      int d = (nt & 3) * 16 + ln;
      int lbase = l0 + m0 + lh * 8;
      v8h hv;
      #pragma unroll
      for (int j = 0; j < 8; ++j) hv[j] = (_Float16)c[j];
      *(v8h*)&Vt[(((size_t)b_idx * NH_ + head) * D_ + d) * L_ + lbase] = hv;
    } else {
      #pragma unroll
      for (int j = 0; j < 8; ++j)
        sStage[wv][(j + lh * 8) * H_ + nt * 16 + ln] = (_Float16)c[j];
    }
  }

  if (proj != 2) {
    asm volatile("s_wait_dscnt 0" ::: "memory"); // wave-private stage; DS is in-order
    _Float16* dst = (proj == 0) ? Qh : Kh;
    #pragma unroll
    for (int i = 0; i < 8; ++i) {
      int cch = lane + i * 32;            // 256 chunks of 8 halfs
      int r = cch >> 4, n0 = (cch & 15) * 8;
      int head = n0 >> 6, d0 = n0 & 63;
      v8h hv = *(const v8h*)&sStage[wv][r * H_ + n0];
      *(v8h*)&dst[(((size_t)b_idx * NH_ + head) * L_ + (l0 + m0 + r)) * D_ + d0] = hv;
    }
  }
}

// ---------------------------------------------------------------------------
// Kernel C: column mean of V per (b,head): Vmean[bh][d] = mean_l Vt[bh][d][l]
// ---------------------------------------------------------------------------
__global__ __launch_bounds__(64)
void vmean_kernel(const _Float16* __restrict__ Vt, float* __restrict__ Vmean)
{
  int bh = blockIdx.x, d = threadIdx.x;
  const _Float16* row = Vt + ((size_t)bh * D_ + d) * L_;
  float s = 0.0f;
  for (int i = 0; i < L_; i += 8) {
    v8h v = *(const v8h*)&row[i];
    #pragma unroll
    for (int j = 0; j < 8; ++j) s += (float)v[j];
  }
  Vmean[bh * D_ + d] = s * (1.0f / (float)L_);
}

// ---------------------------------------------------------------------------
// Kernel B: causal flash attention.  grid=(16 q-tiles, 64 b*head), block=128.
// Each wave owns a 16-query strip; streams k-tiles 0..qt with online softmax.
// K/V tiles are double-buffered in LDS via async global->LDS copies.
// ---------------------------------------------------------------------------
__global__ __launch_bounds__(128)
void attn_kernel(const _Float16* __restrict__ Qh, const _Float16* __restrict__ Kh,
                 const _Float16* __restrict__ Vt, float* __restrict__ out)
{
  __shared__ __align__(16) _Float16 sQ[64 * D_];
  __shared__ __align__(16) _Float16 sK[2][64 * D_];   // ping-pong K tiles
  __shared__ __align__(16) _Float16 sV[2][D_ * 64];   // ping-pong V^T tiles [d][k]
  __shared__ __align__(16) _Float16 sP[4][16 * 64];   // per-wave P stage
  __shared__ __align__(16) float    sO[4][16 * 64];   // per-wave O stage

  const int tid = threadIdx.x;
  const int qt  = blockIdx.x;
  const int bh  = blockIdx.y;
  const int b = bh >> 1, head = bh & 1;

  const _Float16* Qb_ = Qh + (size_t)bh * L_ * D_;
  const _Float16* Kb_ = Kh + (size_t)bh * L_ * D_;
  const _Float16* Vb_ = Vt + (size_t)bh * D_ * L_;

#if USE_ASYNC_LDS
  // kick off K/V tile 0 into buffer 0 (ASYNCcnt-tracked, no VGPR data)
  for (int i = tid; i < 64 * D_ / 8; i += 128) {
    int r = i >> 3, c8 = (i & 7) * 8;
    __builtin_amdgcn_global_load_async_to_lds_b128(
        to_global_v4i(&Kb_[(size_t)r * D_ + c8]), to_lds_v4i(&sK[0][r * D_ + c8]), 0, 0);
    __builtin_amdgcn_global_load_async_to_lds_b128(
        to_global_v4i(&Vb_[(size_t)r * L_ + c8]), to_lds_v4i(&sV[0][r * 64 + c8]), 0, 0);
  }
#endif
  for (int i = tid; i < 64 * D_ / 8; i += 128) {
    int r = i >> 3, c8 = (i & 7) * 8;
    *(v8h*)&sQ[r * D_ + c8] = *(const v8h*)&Qb_[(size_t)(qt * 64 + r) * D_ + c8];
  }
#if USE_ASYNC_LDS
  __builtin_amdgcn_s_wait_asynccnt(0);
#endif
  __syncthreads();

  const int wv = tid >> 5, lane = tid & 31;
  const int lh = lane >> 4, ln = lane & 15;

  Frag16 aq[2];
  #pragma unroll
  for (int kk = 0; kk < 2; ++kk) {
    int base = (wv * 16 + ln) * D_ + kk * 32 + lh * 8;
    aq[kk].h[0] = *(const v8h*)&sQ[base];
    aq[kk].h[1] = *(const v8h*)&sQ[base + 16];
  }

  v8f o[4] = {};
  float mrow[8], lrow[8];
  #pragma unroll
  for (int j = 0; j < 8; ++j) { mrow[j] = -3.0e38f; lrow[j] = 0.0f; }

  for (int kt2 = 0; kt2 <= qt; ++kt2) {
#if USE_ASYNC_LDS
    const int cur = kt2 & 1;
    // prefetch next K/V tile into the other buffer while we compute this one
    if (kt2 < qt) {
      const int nxt = cur ^ 1;
      const int kb = (kt2 + 1) * 64;
      for (int i = tid; i < 64 * D_ / 8; i += 128) {
        int r = i >> 3, c8 = (i & 7) * 8;
        __builtin_amdgcn_global_load_async_to_lds_b128(
            to_global_v4i(&Kb_[(size_t)(kb + r) * D_ + c8]),
            to_lds_v4i(&sK[nxt][r * D_ + c8]), 0, 0);
        __builtin_amdgcn_global_load_async_to_lds_b128(
            to_global_v4i(&Vb_[(size_t)r * L_ + kb + c8]),
            to_lds_v4i(&sV[nxt][r * 64 + c8]), 0, 0);
      }
    }
#else
    const int cur = 0;
    __syncthreads();
    for (int i = tid; i < 64 * D_ / 8; i += 128) {
      int r = i >> 3, c8 = (i & 7) * 8;
      *(v8h*)&sK[0][r * D_ + c8] = *(const v8h*)&Kb_[(size_t)(kt2 * 64 + r) * D_ + c8];
      *(v8h*)&sV[0][r * 64 + c8] = *(const v8h*)&Vb_[(size_t)r * L_ + kt2 * 64 + c8];
    }
    __syncthreads();
#endif

    // S = Q K^T  (16 x 64 strip per wave)
    v8f s[4];
    #pragma unroll
    for (int n4 = 0; n4 < 4; ++n4) {
      v8f c = {};
      #pragma unroll
      for (int kk = 0; kk < 2; ++kk) {
        Frag16 bk;
        int bb = (n4 * 16 + ln) * D_ + kk * 32 + lh * 8;
        bk.h[0] = *(const v8h*)&sK[cur][bb];
        bk.h[1] = *(const v8h*)&sK[cur][bb + 16];
        c = wmma_f32_f16(aq[kk], bk, c);
      }
      s[n4] = c;
    }
    // scale + causal mask (finite NEG -> exp underflows to exact 0)
    const int qg0 = qt * 64 + wv * 16 + lh * 8;
    #pragma unroll
    for (int n4 = 0; n4 < 4; ++n4) {
      int kg = kt2 * 64 + n4 * 16 + ln;
      #pragma unroll
      for (int j = 0; j < 8; ++j) {
        float v = s[n4][j] * 0.125f;
        s[n4][j] = (kg > qg0 + j) ? NEGV : v;
      }
    }
    // online softmax: row spans 16 lanes of one half -> shfl_xor 1,2,4,8
    float alpha[8];
    #pragma unroll
    for (int j = 0; j < 8; ++j) {
      float t = fmaxf(fmaxf(s[0][j], s[1][j]), fmaxf(s[2][j], s[3][j]));
      #pragma unroll
      for (int off = 1; off < 16; off <<= 1) t = fmaxf(t, __shfl_xor(t, off, 32));
      float mn = fmaxf(mrow[j], t);
      alpha[j] = __expf(mrow[j] - mn);
      mrow[j] = mn;
    }
    #pragma unroll
    for (int j = 0; j < 8; ++j) {
      float rs = 0.0f;
      #pragma unroll
      for (int n4 = 0; n4 < 4; ++n4) {
        float p = __expf(s[n4][j] - mrow[j]);
        s[n4][j] = p;
        rs += p;
      }
      #pragma unroll
      for (int off = 1; off < 16; off <<= 1) rs += __shfl_xor(rs, off, 32);
      lrow[j] = lrow[j] * alpha[j] + rs;
      o[0][j] *= alpha[j]; o[1][j] *= alpha[j]; o[2][j] *= alpha[j]; o[3][j] *= alpha[j];
    }
    // P (C-layout) -> LDS -> A-fragment layout (wave-private, DS is in-order)
    #pragma unroll
    for (int n4 = 0; n4 < 4; ++n4)
      #pragma unroll
      for (int j = 0; j < 8; ++j)
        sP[wv][(j + lh * 8) * 64 + n4 * 16 + ln] = (_Float16)s[n4][j];
    asm volatile("s_wait_dscnt 0" ::: "memory");
    Frag16 ap[2];
    #pragma unroll
    for (int kk = 0; kk < 2; ++kk) {
      int base = ln * 64 + kk * 32 + lh * 8;
      ap[kk].h[0] = *(const v8h*)&sP[wv][base];
      ap[kk].h[1] = *(const v8h*)&sP[wv][base + 16];
    }
    // O += P V
    #pragma unroll
    for (int dt = 0; dt < 4; ++dt)
      #pragma unroll
      for (int kk = 0; kk < 2; ++kk) {
        Frag16 bv;
        int bb = (dt * 16 + ln) * 64 + kk * 32 + lh * 8;
        bv.h[0] = *(const v8h*)&sV[cur][bb];
        bv.h[1] = *(const v8h*)&sV[cur][bb + 16];
        o[dt] = wmma_f32_f16(ap[kk], bv, o[dt]);
      }

#if USE_ASYNC_LDS
    if (kt2 < qt) __builtin_amdgcn_s_wait_asynccnt(0);
    __syncthreads();   // next tile fully landed; everyone done with 'cur'
#endif
  }

  #pragma unroll
  for (int j = 0; j < 8; ++j) {
    float inv = 1.0f / lrow[j];
    o[0][j] *= inv; o[1][j] *= inv; o[2][j] *= inv; o[3][j] *= inv;
  }
  #pragma unroll
  for (int dt = 0; dt < 4; ++dt)
    #pragma unroll
    for (int j = 0; j < 8; ++j)
      sO[wv][(j + lh * 8) * 64 + dt * 16 + ln] = o[dt][j];
  asm volatile("s_wait_dscnt 0" ::: "memory");
  #pragma unroll
  for (int i = 0; i < 8; ++i) {
    int cch = lane + i * 32;
    int r = cch >> 4, c4 = (cch & 15) * 4;
    float4 v = *(const float4*)&sO[wv][r * 64 + c4];
    int q = qt * 64 + wv * 16 + r;
    *(float4*)&out[((size_t)b * L_ + q) * H_ + head * D_ + c4] = v;
  }
}

// ---------------------------------------------------------------------------
// Kernel D: masked query rows -> uniform attention over all keys == mean(V)
// ---------------------------------------------------------------------------
__global__ __launch_bounds__(256)
void fixup_kernel(const unsigned char* __restrict__ tmask,
                  const float* __restrict__ Vmean, float* __restrict__ out)
{
  size_t idx = (size_t)blockIdx.x * 256 + threadIdx.x;
  if (idx >= (size_t)B_ * L_ * H_) return;
  int h = (int)(idx & (H_ - 1));
  size_t bq = idx >> 7;
  if (tmask[bq]) {
    int head = h >> 6, d = h & 63;
    int bb = (int)(bq >> 10);
    out[idx] = Vmean[(bb * NH_ + head) * D_ + d];
  }
}

// ---------------------------------------------------------------------------
extern "C" void kernel_launch(void* const* d_in, const int* in_sizes, int n_in,
                              void* d_out, int out_size, void* d_ws, size_t ws_size,
                              hipStream_t stream) {
  (void)in_sizes; (void)n_in; (void)out_size; (void)ws_size;
  const float* queries = (const float*)d_in[0];
  const float* keys    = (const float*)d_in[1];
  const unsigned char* tmask = (const unsigned char*)d_in[2];
  /* d_in[3] = attn_mask: causal triu(k=1), computed analytically */
  const float* Qw = (const float*)d_in[4];
  const float* Qb = (const float*)d_in[5];
  const float* Kw = (const float*)d_in[6];
  const float* Kb = (const float*)d_in[7];
  const float* Vw = (const float*)d_in[8];
  const float* Vb = (const float*)d_in[9];
  float* out = (float*)d_out;

  char* ws = (char*)d_ws;
  _Float16* Qh = (_Float16*)(ws);                              // 8 MB
  _Float16* Kh = (_Float16*)(ws + (size_t)8 * 1024 * 1024);    // 8 MB
  _Float16* Vt = (_Float16*)(ws + (size_t)16 * 1024 * 1024);   // 8 MB
  float*  Vmean = (float*)(ws + (size_t)24 * 1024 * 1024);     // 16 KB

  qkv_proj_kernel<<<dim3(512, 3), 128, 0, stream>>>(queries, keys, Qw, Qb, Kw, Kb, Vw, Vb, Qh, Kh, Vt);
  vmean_kernel<<<dim3(B_ * NH_), 64, 0, stream>>>(Vt, Vmean);
  attn_kernel<<<dim3(L_ / 64, B_ * NH_), 128, 0, stream>>>(Qh, Kh, Vt, out);
  fixup_kernel<<<dim3((B_ * L_ * H_) / 256), 256, 0, stream>>>(tmask, Vmean, out);
}
